// ConvLSTM_22917945491876
// MI455X (gfx1250) — matrix operand order repaired and checked by reading, not compiled, not run on hardware
//
#include <hip/hip_runtime.h>
#include <hip/hip_bf16.h>
#include <math.h>

typedef __attribute__((ext_vector_type(16))) _Float16 v16h;
typedef __attribute__((ext_vector_type(8)))  _Float16 v8h;
typedef __attribute__((ext_vector_type(8)))  float    v8f;

namespace {
constexpr int Tt = 64, Bb = 32, CIN = 64, HID = 128, Ll = 256, LP = 128;
constexpr int NS = Tt * Bb;          // 2048 flattened samples
constexpr float EPS = 1e-5f;
constexpr int KX = CIN * 3;          // 192  (input conv GEMM-K), t-major: k = t*64 + c
constexpr int KH = HID * 3;          // 384  (hidden conv GEMM-K), t-major: k = t*128 + c
constexpr int KX32 = KX / 32;        // 6 K-steps
constexpr int KH32 = KH / 32;        // 12 K-steps
constexpr int XTPITCH = 72;          // LDS pitch (halves) for xsT[258][72]
constexpr int HTPITCH = 136;         // LDS pitch (halves) for hsT[130][136]
constexpr size_t GSTR = (size_t)NS * HID * LP;   // gate stride in Xbuf
}

struct GatePtrs { const float* p[4]; };
struct WPtrs    { const float* wx[4]; const float* wh[4]; };

// ---------------------------------------------------------------------------
// Weight prep: pack f16 A-fragments in WMMA VGPR order.
// Layout: [gate][mtile][k32][lane(32)][j(16)]; per-lane 16 halves contiguous.
// k is t-major (k = t*C + c) to match the transposed activation staging.
// ---------------------------------------------------------------------------
__global__ void prep_weights(WPtrs w, _Float16* __restrict__ wxpk,
                             _Float16* __restrict__ whpk) {
  int idx = blockIdx.x * blockDim.x + threadIdx.x;
  const int NWX = 4 * 8 * KX32 * 32 * 16;   // 98304
  const int NWH = 4 * 8 * KH32 * 32 * 16;   // 196608
  if (idx < NWX) {
    int j = idx & 15, lane = (idx >> 4) & 31;
    int rest = idx >> 9;
    int k32 = rest % KX32; int r2 = rest / KX32;
    int mt = r2 & 7, g = r2 >> 3;
    int half = lane >> 4, ml = lane & 15, m = mt * 16 + ml;
    int p = j >> 1, e = j & 1;
    int kk = half * 16 + (p << 1) + e;
    int k = k32 * 32 + kk;
    int t = k >> 6, c = k & 63;                       // KX: t-major over C=64
    wxpk[idx] = (_Float16)w.wx[g][(m * CIN + c) * 3 + t];
  } else if (idx < NWX + NWH) {
    int i2 = idx - NWX;
    int j = i2 & 15, lane = (i2 >> 4) & 31;
    int rest = i2 >> 9;
    int k32 = rest % KH32; int r2 = rest / KH32;
    int mt = r2 & 7, g = r2 >> 3;
    int half = lane >> 4, ml = lane & 15, m = mt * 16 + ml;
    int p = j >> 1, e = j & 1;
    int kk = half * 16 + (p << 1) + e;
    int k = k32 * 32 + kk;
    int t = k >> 7, c = k & 127;                      // KH: t-major over C=128
    whpk[i2] = (_Float16)w.wh[g][(m * HID + c) * 3 + t];
  }
}

// ---------------------------------------------------------------------------
// Zero c-state and h carry buffer (h0 = 0)
// ---------------------------------------------------------------------------
__global__ void init_state(float* __restrict__ cst, _Float16* __restrict__ htmp) {
  int idx = blockIdx.x * blockDim.x + threadIdx.x;
  if (idx < Bb * HID * LP) { cst[idx] = 0.f; htmp[idx] = (_Float16)0.f; }
}

// ---------------------------------------------------------------------------
// Phase 1: per (sample, gate): conv1d as WMMA GEMM [128x192]x[192x256],
// fused bias + LayerNorm(HID,L) + maxpool2. A from packed global (32B/lane),
// B from transposed LDS (2x ds_load_b128 per fragment).
// ---------------------------------------------------------------------------
__global__ void __launch_bounds__(256)
convx_ln_pool(const float* __restrict__ x, const _Float16* __restrict__ wxpk,
              GatePtrs bx, const float* __restrict__ lnw,
              const float* __restrict__ lnb, float* __restrict__ Xout) {
  extern __shared__ char smem[];
  _Float16* xsT = (_Float16*)smem;                   // [258][XTPITCH], row l+1 = x[:,l]
  float*    red = (float*)(xsT + 258 * XTPITCH);     // [2][256]

  const int sample = blockIdx.x, gate = blockIdx.y, tid = threadIdx.x;

  // stage x transposed with halo rows
  for (int i = tid; i < CIN * Ll; i += 256) {
    int c = i >> 8, l = i & (Ll - 1);
    xsT[(l + 1) * XTPITCH + c] = (_Float16)x[(size_t)sample * (CIN * Ll) + i];
  }
  for (int c = tid; c < CIN; c += 256) {
    xsT[c] = (_Float16)0.f;
    xsT[(Ll + 1) * XTPITCH + c] = (_Float16)0.f;
  }
  __syncthreads();

  const int lane = tid & 31, half = lane >> 4, ml = lane & 15;
  const int mtile = tid >> 5;

  v8f zero = {};
  v8f acc[16];
#pragma unroll
  for (int nt = 0; nt < 16; ++nt) acc[nt] = zero;

  const _Float16* aptr =
      wxpk + ((size_t)((gate * 8 + mtile) * KX32) * 32 + lane) * 16;

  for (int k32 = 0; k32 < KX32; ++k32) {
    v16h a = *(const v16h*)(aptr + (size_t)k32 * (32 * 16));
    __builtin_prefetch(aptr + (size_t)(k32 + 1) * (32 * 16), 0, 1);
    int k0 = k32 * 32 + half * 16;
    int t = k0 >> 6, c0 = k0 & 63;                   // 16-block stays inside one t
    int boff = (ml + t) * XTPITCH + c0;
#pragma unroll
    for (int nt = 0; nt < 16; ++nt) {
      const _Float16* bp = xsT + boff + nt * 16 * XTPITCH;
      v8h blo = *(const v8h*)bp;
      v8h bhi = *(const v8h*)(bp + 8);
      v16h b = __builtin_shufflevector(blo, bhi, 0, 1, 2, 3, 4, 5, 6, 7,
                                       8, 9, 10, 11, 12, 13, 14, 15);
      acc[nt] = __builtin_amdgcn_wmma_f32_16x16x32_f16(
          false, a, false, b, (short)0, acc[nt], false, false);
    }
  }

  // bias + LN statistics
  const float* bxg = bx.p[gate];
  float bias_r[8];
#pragma unroll
  for (int r = 0; r < 8; ++r) bias_r[r] = bxg[mtile * 16 + half * 8 + r];

  float s1 = 0.f, s2 = 0.f;
#pragma unroll
  for (int nt = 0; nt < 16; ++nt)
#pragma unroll
    for (int r = 0; r < 8; ++r) {
      float v = acc[nt][r] + bias_r[r];
      acc[nt][r] = v;
      s1 += v; s2 += v * v;
    }

  red[tid] = s1; red[256 + tid] = s2;
  __syncthreads();
  for (int off = 128; off > 0; off >>= 1) {
    if (tid < off) { red[tid] += red[tid + off]; red[256 + tid] += red[256 + tid + off]; }
    __syncthreads();
  }
  const float inv = 1.f / (float)(HID * Ll);
  float mu  = red[0] * inv;
  float var = red[256] * inv - mu * mu;
  float rs  = rsqrtf(var + EPS);

  // normalize + affine + maxpool2 (pooled pairs are adjacent lanes)
  const size_t obase = ((size_t)gate * NS + sample) * (size_t)(HID * LP);
#pragma unroll
  for (int nt = 0; nt < 16; ++nt) {
#pragma unroll
    for (int r = 0; r < 8; ++r) {
      int m = mtile * 16 + half * 8 + r;
      int n = nt * 16 + ml;
      float v = (acc[nt][r] - mu) * rs * lnw[m * Ll + n] + lnb[m * Ll + n];
      float other = __shfl_xor(v, 1, 32);
      float pm = fmaxf(v, other);
      if ((ml & 1) == 0) Xout[obase + m * LP + (n >> 1)] = pm;
    }
  }
}

// ---------------------------------------------------------------------------
// Phase 2, fully fused per timestep: one block per sample computes the
// [512x384]x[384x128] hidden conv for ALL 4 gates (wave = 16-row m-slice,
// acc[4][8], B fragment shared by the 4 gate WMMAs), then the LSTM pointwise
// update straight from the accumulators. h carried as compact f16.
// ---------------------------------------------------------------------------
__global__ void __launch_bounds__(256)
lstm_step(const _Float16* __restrict__ whpk, GatePtrs bh,
          const float* __restrict__ X, float* __restrict__ cst,
          float* __restrict__ out, _Float16* __restrict__ htmp, int t) {
  extern __shared__ char smem[];
  _Float16* hsT = (_Float16*)smem;                   // [130][HTPITCH], row l+1 = h[:,l]

  const int s = blockIdx.x, tid = threadIdx.x;

  // stage h transposed with halo rows (coalesced global read)
  for (int i = tid; i < HID * LP; i += 256) {
    int m = i >> 7, n = i & 127;
    hsT[(n + 1) * HTPITCH + m] = htmp[(size_t)s * (HID * LP) + i];
  }
  for (int c = tid; c < HID; c += 256) {
    hsT[c] = (_Float16)0.f;
    hsT[(LP + 1) * HTPITCH + c] = (_Float16)0.f;
  }
  __syncthreads();

  const int lane = tid & 31, half = lane >> 4, ml = lane & 15;
  const int w = tid >> 5;                            // wave = m-slice [w*16, w*16+16)

  v8f zero = {};
  v8f acc[4][8];
#pragma unroll
  for (int g = 0; g < 4; ++g)
#pragma unroll
    for (int nt = 0; nt < 8; ++nt) acc[g][nt] = zero;

  for (int k32 = 0; k32 < KH32; ++k32) {
    v16h a[4];
#pragma unroll
    for (int g = 0; g < 4; ++g)
      a[g] = *(const v16h*)(whpk +
              ((size_t)(((g * 8 + w) * KH32 + k32) * 32) + lane) * 16);
    int k0 = k32 * 32 + half * 16;
    int tt = k0 >> 7, c0 = k0 & 127;
    int boff = (ml + tt) * HTPITCH + c0;
#pragma unroll
    for (int nt = 0; nt < 8; ++nt) {
      const _Float16* bp = hsT + boff + nt * 16 * HTPITCH;
      v8h blo = *(const v8h*)bp;
      v8h bhi = *(const v8h*)(bp + 8);
      v16h b = __builtin_shufflevector(blo, bhi, 0, 1, 2, 3, 4, 5, 6, 7,
                                       8, 9, 10, 11, 12, 13, 14, 15);
#pragma unroll
      for (int g = 0; g < 4; ++g)
        acc[g][nt] = __builtin_amdgcn_wmma_f32_16x16x32_f16(
            false, a[g], false, b, (short)0, acc[g][nt], false, false);
    }
  }

  float biasv[4][8];
#pragma unroll
  for (int g = 0; g < 4; ++g)
#pragma unroll
    for (int r = 0; r < 8; ++r) biasv[g][r] = bh.p[g][w * 16 + half * 8 + r];

  const size_t sbase = (size_t)(t * Bb + s) * (HID * LP);
#pragma unroll
  for (int nt = 0; nt < 8; ++nt) {
#pragma unroll
    for (int r = 0; r < 8; ++r) {
      int m = w * 16 + half * 8 + r;
      int n = nt * 16 + ml;
      int ridx = m * LP + n;
      float p0 = acc[0][nt][r] + biasv[0][r] + X[0 * GSTR + sbase + ridx];
      float p1 = acc[1][nt][r] + biasv[1][r] + X[1 * GSTR + sbase + ridx];
      float p2 = acc[2][nt][r] + biasv[2][r] + X[2 * GSTR + sbase + ridx];
      float p3 = acc[3][nt][r] + biasv[3][r] + X[3 * GSTR + sbase + ridx];
      float iv = 1.f / (1.f + __expf(-p0));
      float fv = 1.f / (1.f + __expf(-p1));
      float gv = tanhf(p2);
      float ov = 1.f / (1.f + __expf(-p3));
      size_t ci = (size_t)s * (HID * LP) + ridx;
      float cv = fv * cst[ci] + iv * gv;
      cst[ci] = cv;
      float hv = ov * tanhf(cv);
      out[sbase + ridx] = hv;
      htmp[ci] = (_Float16)hv;   // safe: staging of this sample finished pre-barrier
    }
  }
}

// ---------------------------------------------------------------------------
extern "C" void kernel_launch(void* const* d_in, const int* in_sizes, int n_in,
                              void* d_out, int out_size, void* d_ws, size_t ws_size,
                              hipStream_t stream) {
  WPtrs w; GatePtrs bx, bh;
  const float* x = (const float*)d_in[0];
  for (int g = 0; g < 4; ++g) {
    w.wx[g] = (const float*)d_in[1 + 4 * g];
    bx.p[g] = (const float*)d_in[2 + 4 * g];
    w.wh[g] = (const float*)d_in[3 + 4 * g];
    bh.p[g] = (const float*)d_in[4 + 4 * g];
  }
  const float* lnw = (const float*)d_in[17];
  const float* lnb = (const float*)d_in[18];
  float* out = (float*)d_out;

  // workspace carve (all offsets 256B-aligned)
  char* base = (char*)d_ws;
  float*    Xbuf = (float*)base;
  size_t off = (size_t)4 * GSTR * sizeof(float);                     // 512 MB
  _Float16* wxpk = (_Float16*)(base + off); off += (size_t)4 * 8 * KX32 * 32 * 16 * sizeof(_Float16);
  _Float16* whpk = (_Float16*)(base + off); off += (size_t)4 * 8 * KH32 * 32 * 16 * sizeof(_Float16);
  float*    cst  = (float*)(base + off);    off += (size_t)Bb * HID * LP * sizeof(float);
  _Float16* htmp = (_Float16*)(base + off);

  const int NW = 4 * 8 * KX32 * 32 * 16 + 4 * 8 * KH32 * 32 * 16;
  prep_weights<<<(NW + 255) / 256, 256, 0, stream>>>(w, wxpk, whpk);
  init_state<<<(Bb * HID * LP + 255) / 256, 256, 0, stream>>>(cst, htmp);

  size_t smem1 = (size_t)258 * XTPITCH * sizeof(_Float16) + 512 * sizeof(float);
  convx_ln_pool<<<dim3(NS, 4), 256, smem1, stream>>>(x, wxpk, bx, lnw, lnb, Xbuf);

  size_t smem2 = (size_t)130 * HTPITCH * sizeof(_Float16);
  for (int t = 0; t < Tt; ++t)
    lstm_step<<<Bb, 256, smem2, stream>>>(whpk, bh, Xbuf, cst, out, htmp, t);
}
